// MAPFEncoder_GNN_66048007078000
// MI455X (gfx1250) — matrix operand order repaired and checked by reading, not compiled, not run on hardware
//
#include <hip/hip_runtime.h>
#include <hip/hip_bf16.h>

// ---------------------------------------------------------------------------
// MAPF factorized transformer encoder for gfx1250 (MI455X).
// GEMMs run on v_wmma_f32_16x16x32_bf16 (fp32 accumulate) with a 2x2
// register tile per wave (32x32 output) for ~2x arithmetic intensity.
// Attention is a fused online-softmax fp32 kernel (only ~4% of FLOPs).
// LN / embed / permute are trivially memory-bound elementwise kernels.
// ---------------------------------------------------------------------------

typedef __attribute__((ext_vector_type(16))) __bf16 v16bf;
typedef __attribute__((ext_vector_type(8)))  float  v8f;

#define MTOK   65536    // B*N*T = 16*64*64 tokens
#define DMODEL 256
#define NTOT   16777216 // MTOK * DMODEL

__device__ __forceinline__ unsigned short f2bf(float f) {
  unsigned u = __builtin_bit_cast(unsigned, f);
  u += 0x7FFFu + ((u >> 16) & 1u);           // round-to-nearest-even
  return (unsigned short)(u >> 16);
}
__device__ __forceinline__ float bf2f(unsigned short h) {
  unsigned u = ((unsigned)h) << 16;
  return __builtin_bit_cast(float, u);
}

// ---------------------------------------------------------------------------
// Tiled WMMA GEMM:  out[M,N] = act(A[M,K] @ W[N,K]^T + bias) (+res1) (+res2)
// A, W bf16; fp32 accumulation. Each wave owns a 32x32 output tile computed
// as a 2x2 grid of 16x16 WMMA accumulators (A0/A1 x B0/B1 -> 4 wmma/k-step).
// Block = 256 threads = 8 waves arranged 4(M) x 2(N) -> 128x64 block tile.
// ---------------------------------------------------------------------------
__global__ __launch_bounds__(256) void gemm_bf16_wmma(
    const unsigned short* __restrict__ A,   // [M,K] bf16 row-major
    const unsigned short* __restrict__ W,   // [N,K] bf16 row-major (B = W^T)
    const float* __restrict__ bias,         // [N]
    const float* __restrict__ res1,         // [M,N] fp32 or null
    const float* __restrict__ res2,         // [M,N] fp32 or null
    float* __restrict__ outF,               // [M,N] fp32 or null
    unsigned short* __restrict__ outB,      // [M,N] bf16 or null
    int M, int N, int K, int act)
{
  const int wave = threadIdx.x >> 5;
  const int lane = threadIdx.x & 31;
  const int wm = wave & 3;                   // 4 waves along M
  const int wn = wave >> 2;                  // 2 waves along N
  const int rowBase = blockIdx.x * 128 + wm * 32;
  const int colBase = blockIdx.y * 64 + wn * 32;
  const int r    = lane & 15;
  const int half = lane >> 4;
  (void)M;

  v8f acc00 = {}, acc01 = {}, acc10 = {}, acc11 = {};
  union Frag { v16bf v; unsigned u[8]; };
  Frag a0, a1, b0, b1;

  const unsigned short* Arow0 = A + (size_t)(rowBase + r) * K;
  const unsigned short* Arow1 = Arow0 + (size_t)16 * K;
  const unsigned short* Wrow0 = W + (size_t)(colBase + r) * K;
  const unsigned short* Wrow1 = Wrow0 + (size_t)16 * K;

  for (int k0 = 0; k0 < K; k0 += 32) {
    // A fragment (16x32 bf16): lane r = row; VGPR v holds
    // K = k0 + (v>>2)*16 + half*8 + (v&3)*2 , +1   (ISA 7.12.2 table)
#pragma unroll
    for (int v = 0; v < 8; ++v) {
      const int kk = k0 + ((v >> 2) << 4) + (half << 3) + ((v & 3) << 1);
      a0.u[v] = *(const unsigned*)(Arow0 + kk);
      a1.u[v] = *(const unsigned*)(Arow1 + kk);
    }
    // B fragment (32x16 bf16): lane r = column N; VGPR v packs rows
    // K = k0 + (v>>2)*16 + half*8 + (v&3)*2 , +1 (same order as A).
    const int kb = k0 + (half << 3);
#pragma unroll
    for (int v = 0; v < 8; ++v) {
      b0.u[v] = *(const unsigned*)(Wrow0 + kb + ((v >> 2) << 4) + ((v & 3) << 1));
      b1.u[v] = *(const unsigned*)(Wrow1 + kb + ((v >> 2) << 4) + ((v & 3) << 1));
    }
    acc00 = __builtin_amdgcn_wmma_f32_16x16x32_bf16(false, a0.v, false, b0.v,
                                                    (short)0, acc00, false, false);
    acc01 = __builtin_amdgcn_wmma_f32_16x16x32_bf16(false, a0.v, false, b1.v,
                                                    (short)0, acc01, false, false);
    acc10 = __builtin_amdgcn_wmma_f32_16x16x32_bf16(false, a1.v, false, b0.v,
                                                    (short)0, acc10, false, false);
    acc11 = __builtin_amdgcn_wmma_f32_16x16x32_bf16(false, a1.v, false, b1.v,
                                                    (short)0, acc11, false, false);
  }

  // C/D layout: lane r = col, VGPR v = row v + half*8.
  auto epilogue = [&](const v8f& acc, int rb, int cb) {
    const int col = cb + r;
    const float bi = bias[col];
#pragma unroll
    for (int v = 0; v < 8; ++v) {
      const int row = rb + v + (half << 3);
      float x = acc[v] + bi;
      if (act) x = (x > 0.0f) ? x : 0.01f * x;   // leaky_relu(0.01)
      const size_t idx = (size_t)row * N + col;
      if (res1) x += res1[idx];
      if (res2) x += res2[idx];
      if (outF) outF[idx] = x;
      if (outB) outB[idx] = f2bf(x);
    }
  };
  epilogue(acc00, rowBase,      colBase);
  epilogue(acc01, rowBase,      colBase + 16);
  epilogue(acc10, rowBase + 16, colBase);
  epilogue(acc11, rowBase + 16, colBase + 16);
}

// ---------------------------------------------------------------------------
// LayerNorm over D=256, fp32 in -> bf16 out. One wave per row (8 elems/lane).
// ---------------------------------------------------------------------------
__global__ __launch_bounds__(256) void ln_bf16(
    const float* __restrict__ x, const float* __restrict__ g,
    const float* __restrict__ b, unsigned short* __restrict__ out)
{
  const int row  = blockIdx.x * 8 + (threadIdx.x >> 5);
  const int lane = threadIdx.x & 31;
  const float* xr = x + (size_t)row * DMODEL;
  float v[8], s = 0.0f;
#pragma unroll
  for (int i = 0; i < 8; ++i) { v[i] = xr[lane + (i << 5)]; s += v[i]; }
#pragma unroll
  for (int o = 16; o > 0; o >>= 1) s += __shfl_xor(s, o, 32);
  const float mean = s * (1.0f / DMODEL);
  float q = 0.0f;
#pragma unroll
  for (int i = 0; i < 8; ++i) { float d = v[i] - mean; q += d * d; }
#pragma unroll
  for (int o = 16; o > 0; o >>= 1) q += __shfl_xor(q, o, 32);
  const float inv = rsqrtf(q * (1.0f / DMODEL) + 1e-5f);
  unsigned short* orow = out + (size_t)row * DMODEL;
#pragma unroll
  for (int i = 0; i < 8; ++i) {
    const int c = lane + (i << 5);
    orow[c] = f2bf((v[i] - mean) * inv * g[c] + b[c]);
  }
}

// ---------------------------------------------------------------------------
// Fused attention per (sequence, head): T=64, head_dim=32, online softmax.
// qkv bf16 [seq*64, 768] (q|k|v each 256 wide); out bf16 [seq*64, 256].
// ---------------------------------------------------------------------------
__global__ __launch_bounds__(64) void attn_fused(
    const unsigned short* __restrict__ qkv, unsigned short* __restrict__ out)
{
  __shared__ float Ks[64][32];
  __shared__ float Vs[64][32];
  const int seq  = blockIdx.x;
  const int head = blockIdx.y;
  const int t    = threadIdx.x;
  const unsigned short* qp =
      qkv + ((size_t)seq * 64 + t) * 768 + head * 32;
  float q[32];
#pragma unroll
  for (int c = 0; c < 32; ++c) {
    q[c]     = bf2f(qp[c]);
    Ks[t][c] = bf2f(qp[256 + c]);
    Vs[t][c] = bf2f(qp[512 + c]);
  }
  __syncthreads();

  float m = -1e30f, l = 0.0f, acc[32];
#pragma unroll
  for (int c = 0; c < 32; ++c) acc[c] = 0.0f;
  const float scale = 0.1767766952966369f;   // 1/sqrt(32)
  for (int s = 0; s < 64; ++s) {
    float sc = 0.0f;
#pragma unroll
    for (int c = 0; c < 32; ++c) sc += q[c] * Ks[s][c];
    sc *= scale;
    const float mn   = fmaxf(m, sc);
    const float corr = __expf(m - mn);
    const float p    = __expf(sc - mn);
    l = l * corr + p;
#pragma unroll
    for (int c = 0; c < 32; ++c) acc[c] = acc[c] * corr + p * Vs[s][c];
    m = mn;
  }
  const float inv = 1.0f / l;
  unsigned short* op = out + ((size_t)seq * 64 + t) * 256 + head * 32;
#pragma unroll
  for (int c = 0; c < 32; ++c) op[c] = f2bf(acc[c] * inv);
}

// ---------------------------------------------------------------------------
// Elementwise helpers
// ---------------------------------------------------------------------------
__global__ void wprep_bf16(const float* __restrict__ W,
                           unsigned short* __restrict__ O, int total) {
  const int idx = blockIdx.x * blockDim.x + threadIdx.x;
  if (idx < total) O[idx] = f2bf(W[idx]);
}

__global__ void embed_pe(const int* __restrict__ xin,
                         const float* __restrict__ emb,
                         const float* __restrict__ pe,
                         float* __restrict__ x, int total) {
  const int idx = blockIdx.x * blockDim.x + threadIdx.x;
  if (idx >= total) return;
  const int d  = idx & 255;
  const int t  = (idx >> 8) & 63;
  const int bn = idx >> 14;
  const int tok = xin[bn * 64 + t];
  x[idx] = emb[(size_t)tok * 256 + d] + pe[t * 256 + d];
}

// out[b, i, j, d] = in[b, j, i, d]  (dims 16,64,64,256); self-inverse form.
__global__ void perm_tn(const float* __restrict__ in,
                        float* __restrict__ out, int total) {
  const int idx = blockIdx.x * blockDim.x + threadIdx.x;
  if (idx >= total) return;
  const int d = idx & 255;
  const int j = (idx >> 8) & 63;
  const int i = (idx >> 14) & 63;
  const int b = idx >> 20;
  out[idx] = in[(((size_t)(b * 64 + j) * 64 + i) << 8) + d];
}

// out[bn, d] = x[bn, t=0, d]
__global__ void gather_t0(const float* __restrict__ x,
                          float* __restrict__ out, int total) {
  const int idx = blockIdx.x * blockDim.x + threadIdx.x;
  if (idx >= total) return;
  const int bn = idx >> 8;
  const int d  = idx & 255;
  out[idx] = x[((size_t)bn << 14) + d];
}

// ---------------------------------------------------------------------------
// Host-side encoder block driver
// ---------------------------------------------------------------------------
static void enc_block(hipStream_t s, float* xin, float* yb, float* xout,
                      const float* extraRes,
                      unsigned short* lnbuf, unsigned short* big,
                      const unsigned short* wIn, const unsigned short* wOut,
                      const unsigned short* wL1, const unsigned short* wL2,
                      const float* in_b, const float* out_b,
                      const float* ln1_g, const float* ln1_b,
                      const float* lin1_b, const float* lin2_b,
                      const float* ln2_g, const float* ln2_b)
{
  const int M = MTOK;
  // LN1 -> bf16
  ln_bf16<<<M / 8, 256, 0, s>>>(xin, ln1_g, ln1_b, lnbuf);
  // QKV projection: [M,256] x [768,256]^T -> bf16 qkv in `big`
  gemm_bf16_wmma<<<dim3(M / 128, 768 / 64), 256, 0, s>>>(
      lnbuf, wIn, in_b, nullptr, nullptr, nullptr, big, M, 768, 256, 0);
  // Attention -> bf16 into lnbuf region (ln output already consumed)
  attn_fused<<<dim3(1024, 8), 64, 0, s>>>(big, lnbuf);
  // Output projection + residual(xin) -> fp32 y
  gemm_bf16_wmma<<<dim3(M / 128, 256 / 64), 256, 0, s>>>(
      lnbuf, wOut, out_b, xin, nullptr, yb, nullptr, M, 256, 256, 0);
  // LN2 -> bf16
  ln_bf16<<<M / 8, 256, 0, s>>>(yb, ln2_g, ln2_b, lnbuf);
  // FFN up + leaky_relu -> bf16 h1 in `big`
  gemm_bf16_wmma<<<dim3(M / 128, 1024 / 64), 256, 0, s>>>(
      lnbuf, wL1, lin1_b, nullptr, nullptr, nullptr, big, M, 1024, 256, 1);
  // FFN down + residual(y) [+ extra outer residual for temporal block]
  gemm_bf16_wmma<<<dim3(M / 128, 256 / 64), 256, 0, s>>>(
      big, wL2, lin2_b, yb, extraRes, xout, nullptr, M, 256, 1024, 0);
}

extern "C" void kernel_launch(void* const* d_in, const int* in_sizes, int n_in,
                              void* d_out, int out_size, void* d_ws, size_t ws_size,
                              hipStream_t stream) {
  (void)in_sizes; (void)n_in; (void)out_size; (void)ws_size;

  const int*   x_in = (const int*)d_in[0];
  // d_in[1] = map_in: unused by the reference forward pass
  const float* emb  = (const float*)d_in[2];
  const float* pe   = (const float*)d_in[3];
  const float* tp[12];
  const float* sp[12];
  for (int j = 0; j < 12; ++j) {
    tp[j] = (const float*)d_in[4 + j];
    sp[j] = (const float*)d_in[16 + j];
  }
  // _PNAMES: 0 in_w,1 in_b,2 out_w,3 out_b,4 ln1_g,5 ln1_b,
  //          6 lin1_w,7 lin1_b,8 lin2_w,9 lin2_b,10 ln2_g,11 ln2_b

  char* ws = (char*)d_ws;
  size_t off = 0;
  auto alloc = [&](size_t bytes) {
    size_t o = off; off += (bytes + 255) & ~(size_t)255; return o;
  };
  float* regA = (float*)(ws + alloc((size_t)NTOT * 4));          // x
  float* regB = (float*)(ws + alloc((size_t)NTOT * 4));          // y_t / xs
  unsigned short* bfS = (unsigned short*)(ws + alloc((size_t)NTOT * 2)); // ln/attn
  unsigned short* big = (unsigned short*)(ws + alloc((size_t)MTOK * 1024 * 2)); // qkv/h1
  unsigned short* wInT  = (unsigned short*)(ws + alloc((size_t)3 * 768 * 256 * 2));
  unsigned short* wOutT = (unsigned short*)(ws + alloc((size_t)3 * 256 * 256 * 2));
  unsigned short* wL1T  = (unsigned short*)(ws + alloc((size_t)3 * 1024 * 256 * 2));
  unsigned short* wL2T  = (unsigned short*)(ws + alloc((size_t)3 * 256 * 1024 * 2));
  unsigned short* wInS  = (unsigned short*)(ws + alloc((size_t)3 * 768 * 256 * 2));
  unsigned short* wOutS = (unsigned short*)(ws + alloc((size_t)3 * 256 * 256 * 2));
  unsigned short* wL1S  = (unsigned short*)(ws + alloc((size_t)3 * 1024 * 256 * 2));
  unsigned short* wL2S  = (unsigned short*)(ws + alloc((size_t)3 * 256 * 1024 * 2));

  // ---- weight fp32 -> bf16 (layout kept [N,K]; K-pairs contiguous for WMMA B)
  auto prep = [&](const float* W, unsigned short* O, int total) {
    wprep_bf16<<<(total + 255) / 256, 256, 0, stream>>>(W, O, total);
  };
  prep(tp[0], wInT,  3 * 768 * 256);
  prep(tp[2], wOutT, 3 * 256 * 256);
  prep(tp[6], wL1T,  3 * 1024 * 256);
  prep(tp[8], wL2T,  3 * 256 * 1024);
  prep(sp[0], wInS,  3 * 768 * 256);
  prep(sp[2], wOutS, 3 * 256 * 256);
  prep(sp[6], wL1S,  3 * 1024 * 256);
  prep(sp[8], wL2S,  3 * 256 * 1024);

  // ---- embedding gather + positional encoding -> x (fp32)
  embed_pe<<<(NTOT + 255) / 256, 256, 0, stream>>>(x_in, emb, pe, regA, NTOT);

  for (int i = 0; i < 3; ++i) {
    // Temporal block: x = x + enc(x)  (extra outer residual = regA)
    enc_block(stream, regA, regB, regA, regA, bfS, big,
              wInT + (size_t)i * 768 * 256, wOutT + (size_t)i * 256 * 256,
              wL1T + (size_t)i * 1024 * 256, wL2T + (size_t)i * 256 * 1024,
              tp[1] + i * 768, tp[3] + i * 256, tp[4] + i * 256, tp[5] + i * 256,
              tp[7] + i * 1024, tp[9] + i * 256, tp[10] + i * 256, tp[11] + i * 256);
    // [bn, t, d] -> [bt, n, d]
    perm_tn<<<(NTOT + 255) / 256, 256, 0, stream>>>(regA, regB, NTOT);
    // Spatial block: xs = enc(xs)
    enc_block(stream, regB, regA, regB, nullptr, bfS, big,
              wInS + (size_t)i * 768 * 256, wOutS + (size_t)i * 256 * 256,
              wL1S + (size_t)i * 1024 * 256, wL2S + (size_t)i * 256 * 1024,
              sp[1] + i * 768, sp[3] + i * 256, sp[4] + i * 256, sp[5] + i * 256,
              sp[7] + i * 1024, sp[9] + i * 256, sp[10] + i * 256, sp[11] + i * 256);
    // [bt, n, d] -> [bn, t, d]
    perm_tn<<<(NTOT + 255) / 256, 256, 0, stream>>>(regB, regA, NTOT);
  }

  // Output: x[:, :, t=0] -> [B, N, D]
  gather_t0<<<(262144 + 255) / 256, 256, 0, stream>>>(regA, (float*)d_out, 262144);
}
